// RBMSites_31688268710512
// MI455X (gfx1250) — matrix operand order repaired and checked by reading, not compiled, not run on hardware
//
#include <hip/hip_runtime.h>
#include <math.h>
#include <stdint.h>

#define NV     96         // visible sites (GEMM reduction dim, and site loop)
#define NH     192        // hidden units
#define BATCHN 1024
#define TRI    4656       // NV*(NV+1)/2
#define LDA    104        // padded LDS row length in halves (208B -> conflict-free b128)
#define BT     64         // batch tile per block
#define MT     12         // 192/16 m-tiles

typedef _Float16 v16h __attribute__((ext_vector_type(16)));
typedef _Float16 h8   __attribute__((ext_vector_type(8)));
typedef float    v8f  __attribute__((ext_vector_type(8)));

union Frag { v16h v; h8 h[2]; };

typedef __attribute__((address_space(3))) const void lds_cvoid;

// LDS byte offset of a __shared__ object (generic -> AS3 -> u32)
__device__ __forceinline__ uint32_t lds_off(const void* p) {
    return (uint32_t)(uintptr_t)(lds_cvoid*)p;
}

// Async 16B copy: global -> LDS, tracked by ASYNCcnt (CDNA5 path)
__device__ __forceinline__ void async_copy16(uint32_t lds_addr, const void* gaddr) {
    asm volatile("global_load_async_to_lds_b128 %0, %1, off"
                 :: "v"(lds_addr), "v"((uint64_t)(uintptr_t)gaddr)
                 : "memory");
}

__device__ __forceinline__ void wait_async0() {
    asm volatile("s_wait_asynccnt 0x0" ::: "memory");
}

// ---------------- prep: expand strict-lower-tri weights into dense f16 A ----
__global__ __launch_bounds__(256)
void prep_wexp(const float* __restrict__ w, _Float16* __restrict__ wexp) {
    int e = blockIdx.x * 256 + threadIdx.x;
    if (e >= NV * NH * NV) return;
    int j = e % NV;
    int h = (e / NV) % NH;
    int k = e / (NV * NH);
    float v = 0.0f;
    if (j < k) v = w[(size_t)h * TRI + (k * (k + 1)) / 2 + j];
    wexp[e] = (_Float16)v;
}

__global__ __launch_bounds__(256)
void prep_x(const float* __restrict__ x, _Float16* __restrict__ xh) {
    int e = blockIdx.x * 256 + threadIdx.x;
    if (e < BATCHN * NV) xh[e] = (_Float16)x[e];
}

// ---------------- main fused GEMM (WMMA) + cos-product epilogue -------------
__global__ __launch_bounds__(256)
void rbm_site_kernel(const _Float16* __restrict__ wexp,
                     const _Float16* __restrict__ xh,
                     const float* __restrict__ x,
                     const float* __restrict__ w,
                     const float* __restrict__ hb,
                     float* __restrict__ probs) {
    const int k     = blockIdx.y;          // site
    const int bbase = blockIdx.x * BT;     // batch tile base
    const int tid   = threadIdx.x;
    const int lane  = tid & 31;
    const int wave  = tid >> 5;

    __shared__ __align__(16) _Float16 As[NH * LDA];  // 192 x 96 (padded)
    __shared__ __align__(16) _Float16 Bs[BT * LDA];  // 64  x 96 (padded)
    __shared__ float hbs[NH];
    __shared__ float dgs[NH];
    __shared__ float red[8][32][2];

    const uint32_t asBase = lds_off(As);
    const uint32_t bsBase = lds_off(Bs);

    // Stage A tile (site k's expanded weights) via async global->LDS b128:
    // 192 rows x 12 x 16B chunks = 2304 chunks (9 per thread)
    const _Float16* wk = wexp + (size_t)k * NH * NV;
    for (int c = tid; c < NH * 12; c += 256) {
        int row = c / 12, col = (c % 12) * 8;
        async_copy16(asBase + (uint32_t)(row * LDA + col) * 2,
                     &wk[row * NV + col]);
    }
    // Stage B tile (x^T for this batch tile): 64 rows x 12 chunks
    for (int c = tid; c < BT * 12; c += 256) {
        int row = c / 12, col = (c % 12) * 8;
        async_copy16(bsBase + (uint32_t)(row * LDA + col) * 2,
                     &xh[(size_t)(bbase + row) * NV + col]);
    }
    if (tid < NH) {
        hbs[tid] = hb[tid];
        dgs[tid] = w[(size_t)tid * TRI + (k * (k + 1)) / 2 + k];  // diagonal weight
    }
    wait_async0();
    __syncthreads();

    // wave -> (ntile in 0..3, mhalf in 0..1); each wave owns 6 m-tiles
    const int ntile = wave & 3;
    const int mhalf = wave >> 2;
    const int lrow  = lane & 15;
    const int lhi   = lane >> 4;

    v8f acc[6] = {};

    #pragma unroll
    for (int js = 0; js < 3; ++js) {              // 96 = 3 x 32 reduction steps
        const int jb = js * 32;
        // Load ALL fragments for this K-step first (distinct regs), then issue
        // the 6 WMMAs back-to-back: one dscnt wait, no per-WMMA hazard NOPs.
        Frag bf;
        {   // B frag: lane holds col n=lrow, K chunks at lhi*8 and lhi*8+16
            const _Float16* bp = &Bs[(ntile * 16 + lrow) * LDA + jb + lhi * 8];
            bf.h[0] = *(const h8*)(bp);
            bf.h[1] = *(const h8*)(bp + 16);
        }
        Frag af[6];
        #pragma unroll
        for (int i = 0; i < 6; ++i) {
            const int mt = mhalf * 6 + i;
            // A frag: lane holds row m=lrow, K chunks at lhi*8 and lhi*8+16
            const _Float16* ap = &As[(mt * 16 + lrow) * LDA + jb + lhi * 8];
            af[i].h[0] = *(const h8*)(ap);
            af[i].h[1] = *(const h8*)(ap + 16);
        }
        #pragma unroll
        for (int i = 0; i < 6; ++i) {
            acc[i] = __builtin_amdgcn_wmma_f32_16x16x32_f16(
                false, af[i].v, false, bf.v, (short)0, acc[i], false, false);
        }
    }

    // Epilogue: product over this lane's h values of cos(theta).
    // (The reference's 2^NH factor cancels in val1/||val||, so we accumulate
    //  prod cos(theta) which stays ~1 and never overflows.)
    float p0 = 1.0f, p1 = 1.0f;
    #pragma unroll
    for (int i = 0; i < 6; ++i) {
        const int mt = mhalf * 6 + i;
        #pragma unroll
        for (int r = 0; r < 8; ++r) {
            int hrow = mt * 16 + lhi * 8 + r;     // C/D layout: M = lhi*8 + r
            float t0 = hbs[hrow] + acc[i][r];     // channel without diag
            float t1 = t0 + dgs[hrow];            // channel c == bit (adds diag)
            p0 *= __cosf(t0);
            p1 *= __cosf(t1);
        }
    }
    red[wave][lane][0] = p0;
    red[wave][lane][1] = p1;
    __syncthreads();

    // Combine: for batch col n, product of 4 partials (lane halves x wave pair)
    if (tid < BT) {
        int nt = tid >> 4, l = tid & 15;
        float cn = red[nt][l][0] * red[nt][l + 16][0] *
                   red[nt + 4][l][0] * red[nt + 4][l + 16][0];
        float cd = red[nt][l][1] * red[nt][l + 16][1] *
                   red[nt + 4][l][1] * red[nt + 4][l + 16][1];
        float xv = x[(size_t)(bbase + tid) * NV + k];
        float v1 = (xv > 0.0f) ? cd : cn;         // channel c=1
        float v0 = (xv > 0.0f) ? cn : cd;         // channel c=0
        float nrm = sqrtf(v0 * v0 + v1 * v1);
        probs[(size_t)k * BATCHN + bbase + tid] = v1 / fmaxf(nrm, 1e-30f);
    }
}

// ---------------- final product over sites ----------------------------------
__global__ __launch_bounds__(256)
void reduce_kernel(const float* __restrict__ probs, float* __restrict__ out) {
    int b = blockIdx.x * 256 + threadIdx.x;
    if (b >= BATCHN) return;
    float p = 1.0f;
    #pragma unroll 4
    for (int k = 0; k < NV; ++k) p *= probs[(size_t)k * BATCHN + b];
    out[b] = p;
}

extern "C" void kernel_launch(void* const* d_in, const int* in_sizes, int n_in,
                              void* d_out, int out_size, void* d_ws, size_t ws_size,
                              hipStream_t stream) {
    (void)in_sizes; (void)n_in; (void)out_size; (void)ws_size;
    const float* x  = (const float*)d_in[0];
    const float* w  = (const float*)d_in[1];
    const float* hb = (const float*)d_in[2];
    float* out = (float*)d_out;

    char* ws = (char*)d_ws;
    _Float16* wexp = (_Float16*)ws;                                   // 3.46 MB
    _Float16* xh   = (_Float16*)(ws + (size_t)NV * NH * NV * 2);      // 192 KB
    float*    prb  = (float*)(ws + (size_t)NV * NH * NV * 2
                                 + (size_t)BATCHN * NV * 2);          // 384 KB

    prep_wexp<<<(NV * NH * NV + 255) / 256, 256, 0, stream>>>(w, wexp);
    prep_x<<<(BATCHN * NV + 255) / 256, 256, 0, stream>>>(x, xh);

    dim3 grid(BATCHN / BT, NV);   // 16 x 96 = 1536 workgroups
    rbm_site_kernel<<<grid, 256, 0, stream>>>(wexp, xh, x, w, hb, prb);

    reduce_kernel<<<(BATCHN + 255) / 256, 256, 0, stream>>>(prb, out);
}
